// GraphAttentionLayer_41437844472352
// MI455X (gfx1250) — matrix operand order repaired and checked by reading, not compiled, not run on hardware
//
#include <hip/hip_runtime.h>

// GAT layer, B=4, N=4096, FIN=256, FOUT=64.
// Key identity: the attention logit is constant in j, so softmax collapses to
// att[b,i,j] = adj[b,i,j] / rowcount_i  (a-vector and leakyrelu are dead code).
// out = elu( (adj_row . Wh) / cnt ),  Wh = h @ W.
// adj streaming (256 MB) is the roofline bottleneck (~11us at 23.3 TB/s);
// matmuls run on v_wmma_f32_16x16x32_f16 (0/1 adj is exact in f16).

#define B_    4
#define N_    4096
#define FIN_  256
#define FOUT_ 64

typedef __attribute__((ext_vector_type(16))) _Float16     v16h;
typedef __attribute__((ext_vector_type(8)))  _Float16     v8h;
typedef __attribute__((ext_vector_type(8)))  float        v8f;
typedef __attribute__((ext_vector_type(4)))  float        v4f;
typedef __attribute__((ext_vector_type(4)))  unsigned int v4u;

// ---------------------------------------------------------------------------
// Kernel 0: W [FIN][FOUT] f32  ->  Wt [FOUT][FIN] f16 (transpose + convert)
// ---------------------------------------------------------------------------
__global__ void gat_prep_w(const float* __restrict__ W, _Float16* __restrict__ Wt) {
  int idx = blockIdx.x * blockDim.x + threadIdx.x;
  if (idx >= FIN_ * FOUT_) return;
  int f = idx & (FOUT_ - 1);
  int k = idx >> 6;
  Wt[f * FIN_ + k] = (_Float16)W[k * FOUT_ + f];
}

// ---------------------------------------------------------------------------
// Kernel 1: Wh = h @ W, stored transposed+f16:  WhT[b][f][m], f in [0,64),
// m in [0,4096). Wave = one M16 strip x F64. WMMA f16, f32 accumulate.
// A-operand (16x32 f16): lane lo<16 holds row m0+lo, K = kb..kb+15 (v0-7),
// kb = k0 + hi*16 (contiguous).  B-operand (32x16 f16): lane
// holds column f, K = k0 + hi*16 .. +15, contiguous from Wt.
// ---------------------------------------------------------------------------
__global__ void gat_gemm1(const float* __restrict__ h,
                          const _Float16* __restrict__ Wt,
                          _Float16* __restrict__ WhT) {
  const int lane = threadIdx.x & 31;
  const int wave = threadIdx.x >> 5;
  const int lo   = lane & 15;
  const int hi   = lane >> 4;
  const int b    = blockIdx.x >> 5;
  const int m0   = (blockIdx.x & 31) * 128 + wave * 16;
  const int row  = m0 + lo;

  const float* hrow = h + ((size_t)b * N_ + row) * FIN_;

  v8f acc[4] = {};

  for (int k0 = 0; k0 < FIN_; k0 += 32) {
    const int kb = k0 + hi * 16;
    v4f x0 = *(const v4f*)(hrow + kb);
    v4f x1 = *(const v4f*)(hrow + kb + 4);
    v4f x2 = *(const v4f*)(hrow + kb + 8);
    v4f x3 = *(const v4f*)(hrow + kb + 12);
    v16h A;
#pragma unroll
    for (int i = 0; i < 4; ++i) {
      A[i]      = (_Float16)x0[i];
      A[4 + i]  = (_Float16)x1[i];
      A[8 + i]  = (_Float16)x2[i];
      A[12 + i] = (_Float16)x3[i];
    }
    const _Float16* bp = Wt + (size_t)lo * FIN_ + k0 + hi * 16;
#pragma unroll
    for (int n = 0; n < 4; ++n) {
      v16h Bm = *(const v16h*)(bp + (size_t)n * 16 * FIN_);
      acc[n] = __builtin_amdgcn_wmma_f32_16x16x32_f16(
          false, A, false, Bm, (short)0, acc[n], false, false);
    }
  }

  // C/D layout: VGPR r, lane lo (hi=0) -> M = r, hi=1 -> M = 8+r; N = lo.
  // Store 8 consecutive rows (contiguous in WhT's m dimension) as one b128.
#pragma unroll
  for (int n = 0; n < 4; ++n) {
    v8h o;
#pragma unroll
    for (int r = 0; r < 8; ++r) o[r] = (_Float16)acc[n][r];
    _Float16* dst = WhT + (size_t)(b * FOUT_ + n * 16 + lo) * N_ + m0 + hi * 8;
    *(v8h*)dst = o;
  }
}

// ---------------------------------------------------------------------------
// Kernel 2: out[b][i][f] = elu( (adj_row_i . Wh[:,f]) / cnt_i ).
// A = adj tile (int 0/1 -> f16 0x3C00/0, exact). 5th WMMA vs all-ones B
// accumulates cnt_i directly in C-layout (perfect slot for the epilogue scale).
// ---------------------------------------------------------------------------
__global__ void gat_gemm2(const int* __restrict__ adj,
                          const _Float16* __restrict__ WhT,
                          float* __restrict__ out) {
  const int lane = threadIdx.x & 31;
  const int wave = threadIdx.x >> 5;
  const int lo   = lane & 15;
  const int hi   = lane >> 4;
  const int b    = blockIdx.x >> 5;
  const int m0   = (blockIdx.x & 31) * 128 + wave * 16;
  const int row  = m0 + lo;

  const int*      arow  = adj + ((size_t)b * N_ + row) * N_;
  const _Float16* bbase = WhT + (size_t)(b * FOUT_ + lo) * N_;

  v16h ones;
#pragma unroll
  for (int i = 0; i < 16; ++i) ones[i] = (_Float16)1.0f;

  v8f acc[4] = {};
  v8f accC   = {};

  for (int k0 = 0; k0 < N_; k0 += 32) {
    const int kb = k0 + hi * 16;
    if (k0 + 1056 <= N_) __builtin_prefetch(arow + kb + 1024, 0, 0);

    v4u i0 = *(const v4u*)(arow + kb);
    v4u i1 = *(const v4u*)(arow + kb + 4);
    v4u i2 = *(const v4u*)(arow + kb + 8);
    v4u i3 = *(const v4u*)(arow + kb + 12);

    union { v16h h16; unsigned u[8]; } Au;
    Au.u[0] = (i0[0] ? 0x3C00u : 0u) | (i0[1] ? 0x3C000000u : 0u);
    Au.u[1] = (i0[2] ? 0x3C00u : 0u) | (i0[3] ? 0x3C000000u : 0u);
    Au.u[2] = (i1[0] ? 0x3C00u : 0u) | (i1[1] ? 0x3C000000u : 0u);
    Au.u[3] = (i1[2] ? 0x3C00u : 0u) | (i1[3] ? 0x3C000000u : 0u);
    Au.u[4] = (i2[0] ? 0x3C00u : 0u) | (i2[1] ? 0x3C000000u : 0u);
    Au.u[5] = (i2[2] ? 0x3C00u : 0u) | (i2[3] ? 0x3C000000u : 0u);
    Au.u[6] = (i3[0] ? 0x3C00u : 0u) | (i3[1] ? 0x3C000000u : 0u);
    Au.u[7] = (i3[2] ? 0x3C00u : 0u) | (i3[3] ? 0x3C000000u : 0u);
    v16h A = Au.h16;

    const _Float16* bp = bbase + k0 + hi * 16;
#pragma unroll
    for (int n = 0; n < 4; ++n) {
      v16h Bm = *(const v16h*)(bp + (size_t)n * 16 * N_);
      acc[n] = __builtin_amdgcn_wmma_f32_16x16x32_f16(
          false, A, false, Bm, (short)0, acc[n], false, false);
    }
    accC = __builtin_amdgcn_wmma_f32_16x16x32_f16(
        false, A, false, ones, (short)0, accC, false, false);
  }

  // accC[r] holds cnt for row (m0 + hi*8 + r) — exactly the row acc[n][r]
  // belongs to. (cnt==0 is probability ~0 for Bernoulli(0.5) rows; guard to 0.)
  float inv[8];
#pragma unroll
  for (int r = 0; r < 8; ++r) {
    float c = accC[r];
    inv[r] = (c > 0.f) ? (1.f / c) : 0.f;
  }

#pragma unroll
  for (int n = 0; n < 4; ++n) {
    float* op = out + ((size_t)b * N_ + m0 + hi * 8) * FOUT_ + n * 16 + lo;
#pragma unroll
    for (int r = 0; r < 8; ++r) {
      float v = acc[n][r] * inv[r];
      v = (v > 0.f) ? v : (__expf(v) - 1.f);   // ELU, alpha = 1
      op[(size_t)r * FOUT_] = v;
    }
  }
}

// ---------------------------------------------------------------------------
extern "C" void kernel_launch(void* const* d_in, const int* in_sizes, int n_in,
                              void* d_out, int out_size, void* d_ws, size_t ws_size,
                              hipStream_t stream) {
  const float* h   = (const float*)d_in[0];
  const int*   adj = (const int*)d_in[1];
  const float* W   = (const float*)d_in[2];
  // d_in[3] (a) is provably unused: the softmax over a j-constant logit
  // collapses to adjacency-row averaging.
  float* out = (float*)d_out;

  _Float16* Wt  = (_Float16*)d_ws;                          // 32 KB
  _Float16* WhT = (_Float16*)((char*)d_ws + 65536);         // 2 MB

  gat_prep_w<<<dim3((FIN_ * FOUT_ + 255) / 256), dim3(256), 0, stream>>>(W, Wt);
  gat_gemm1<<<dim3(B_ * (N_ / 128)), dim3(256), 0, stream>>>(h, Wt, WhT);
  gat_gemm2<<<dim3(B_ * (N_ / 128)), dim3(256), 0, stream>>>(adj, WhT, out);
}
